// Node2Edge2NodeBlockSingleLevel_26250840113773
// MI455X (gfx1250) — matrix operand order repaired and checked by reading, not compiled, not run on hardware
//
#include <hip/hip_runtime.h>
#include <hip/hip_bf16.h>

typedef __attribute__((ext_vector_type(16))) __bf16 v16bf;
typedef __attribute__((ext_vector_type(8)))  float  v8f;

#define D_DIM 128
#define B_SZ  4
#define NS_SZ 10000
#define NT_SZ 10000
#define E_SZ  160000
#define K_SZ  16

__device__ __forceinline__ float silu_f(float x) {
  return x / (1.0f + __expf(-x));
}

// A-fragment (16x32 bf16, MxK): lane = row M (both halves), halves split K 0-7 / 8-15,
// VGPRs 4-7 hold K+16.  Source rows are f32, converted to bf16 on the fly.
__device__ __forceinline__ v16bf load_a_frag_global(const float* __restrict__ row_ptr,
                                                    int lane, int kt) {
  v16bf a;
  const int khalf = (lane >> 4) << 3;          // 0 or 8
  const float* p0 = row_ptr + kt * 32 + khalf; // K = kt*32 + khalf + j
  const float* p1 = p0 + 16;                   // K = kt*32 + 16 + khalf + j
#pragma unroll
  for (int j = 0; j < 8; ++j) {
    a[j]     = (__bf16)p0[j];
    a[j + 8] = (__bf16)p1[j];
  }
  return a;
}

// B-fragment (32x16 bf16, KxN): N = lane&15, K = 8*(lane>>4) + i (i<8), K+16 for i>=8.
// Weight is staged transposed in LDS (wT[n*128 + k]) so this is two contiguous 16B reads.
__device__ __forceinline__ v16bf load_b_frag_lds(const __bf16* wT, int lane, int nt, int kt) {
  const int n  = nt * 16 + (lane & 15);
  const int k0 = kt * 32 + ((lane >> 4) << 3);
  const __bf16* p = wT + n * D_DIM + k0;
  v16bf b;
#pragma unroll
  for (int j = 0; j < 8; ++j) {
    b[j]     = p[j];
    b[j + 8] = p[j + 16];
  }
  return b;
}

// Cooperative f32 -> bf16 transpose stage of a 128x128 weight into LDS.
__device__ __forceinline__ void stage_weight_bf16(const float* __restrict__ W, __bf16* wT) {
  for (int idx = threadIdx.x; idx < D_DIM * D_DIM; idx += blockDim.x) {
    const int k = idx >> 7;
    const int n = idx & 127;
    wT[n * D_DIM + k] = (__bf16)W[idx];
  }
}

// ---------------------------------------------------------------------------
// Kernel 1: P = X @ W   (X: [nrows, 128] f32, W: [128,128] f32, P f32)
// ---------------------------------------------------------------------------
__global__ __launch_bounds__(256, 2)
void proj_gemm_kernel(const float* __restrict__ X, const float* __restrict__ W,
                      float* __restrict__ P, int nrows) {
  __shared__ __attribute__((aligned(32))) __bf16 wT[D_DIM * D_DIM];
  stage_weight_bf16(W, wT);
  __syncthreads();
  const int lane   = threadIdx.x & 31;
  const int wave   = threadIdx.x >> 5;
  const int wpb    = blockDim.x >> 5;
  const int ntiles = nrows >> 4;
  for (int tile = blockIdx.x * wpb + wave; tile < ntiles; tile += gridDim.x * wpb) {
    const float* xrow = X + (size_t)(tile * 16 + (lane & 15)) * D_DIM;
    v16bf a[4];
#pragma unroll
    for (int kt = 0; kt < 4; ++kt) a[kt] = load_a_frag_global(xrow, lane, kt);
    v8f acc[8] = {};
    // kt outer / nt inner: consecutive WMMAs hit different accumulators (no RAW chain)
#pragma unroll
    for (int kt = 0; kt < 4; ++kt) {
      v16bf bw[8];
#pragma unroll
      for (int nt = 0; nt < 8; ++nt) bw[nt] = load_b_frag_lds(wT, lane, nt, kt);
#pragma unroll
      for (int nt = 0; nt < 8; ++nt)
        acc[nt] = __builtin_amdgcn_wmma_f32_16x16x32_bf16(false, a[kt], false, bw[nt],
                                                          (short)0, acc[nt], false, false);
    }
    // row-wise store: one base address per row, nt*64B folds into imm offsets
#pragma unroll
    for (int j = 0; j < 8; ++j) {
      const int m = ((lane >> 4) << 3) + j;
      float* pr = P + (size_t)(tile * 16 + m) * D_DIM + (lane & 15);
#pragma unroll
      for (int nt = 0; nt < 8; ++nt) pr[nt * 16] = acc[nt][j];
    }
  }
}

// ---------------------------------------------------------------------------
// Kernel 2: d_bond = LN(silu(bond@W_e2e + gather(P_s) + gather(P_t)))
//           out0   = bond + d_bond
// One 16x128 tile per wave, full tile kept in 64 accumulator VGPRs.
// ---------------------------------------------------------------------------
__global__ __launch_bounds__(256, 2)
void edge_kernel(const float* __restrict__ bond,
                 const float* __restrict__ Ps,
                 const float* __restrict__ Pt,
                 const int*   __restrict__ src_order,
                 const int*   __restrict__ tgt_order,
                 const float* __restrict__ W_e2e,
                 const float* __restrict__ ln_g,
                 const float* __restrict__ ln_b,
                 float* __restrict__ d_bond,
                 float* __restrict__ out0) {
  __shared__ __attribute__((aligned(32))) __bf16 wT[D_DIM * D_DIM];
  stage_weight_bf16(W_e2e, wT);
  __syncthreads();
  const int lane   = threadIdx.x & 31;
  const int wave   = threadIdx.x >> 5;
  const int wpb    = blockDim.x >> 5;
  const int ntiles = (B_SZ * E_SZ) >> 4;
  float g[8], be[8];
#pragma unroll
  for (int nt = 0; nt < 8; ++nt) {
    g[nt]  = ln_g[nt * 16 + (lane & 15)];
    be[nt] = ln_b[nt * 16 + (lane & 15)];
  }
  const int tstep = gridDim.x * wpb;
  for (int tile = blockIdx.x * wpb + wave; tile < ntiles; tile += tstep) {
    const int r0 = tile << 4;
    const int b  = r0 / E_SZ;          // tiles never cross batches (E % 16 == 0)
    const int e0 = r0 - b * E_SZ;
    if (tile + tstep < ntiles)         // hint next tile into cache
      __builtin_prefetch(bond + (size_t)((tile + tstep) * 16 + (lane & 15)) * D_DIM, 0, 1);
    const float* brow = bond + (size_t)(r0 + (lane & 15)) * D_DIM;
    v16bf a[4];
#pragma unroll
    for (int kt = 0; kt < 4; ++kt) a[kt] = load_a_frag_global(brow, lane, kt);
    v8f acc[8] = {};
#pragma unroll
    for (int kt = 0; kt < 4; ++kt) {
      v16bf bw[8];
#pragma unroll
      for (int nt = 0; nt < 8; ++nt) bw[nt] = load_b_frag_lds(wT, lane, nt, kt);
#pragma unroll
      for (int nt = 0; nt < 8; ++nt)
        acc[nt] = __builtin_amdgcn_wmma_f32_16x16x32_bf16(false, a[kt], false, bw[nt],
                                                          (short)0, acc[nt], false, false);
    }
    float mean[8], rstd[8];
#pragma unroll
    for (int j = 0; j < 8; ++j) {
      const int m = ((lane >> 4) << 3) + j;           // C-layout row
      const int e = e0 + m;
      const int sidx = src_order[e];
      const int tidx = tgt_order[e];
      const float* psrow = Ps + ((size_t)b * NS_SZ + sidx) * D_DIM + (lane & 15);
      const float* ptrow = Pt + ((size_t)b * NT_SZ + tidx) * D_DIM + (lane & 15);
      float s = 0.f, s2 = 0.f;
#pragma unroll
      for (int nt = 0; nt < 8; ++nt) {
        float v = acc[nt][j] + psrow[nt * 16] + ptrow[nt * 16];
        v = silu_f(v);
        acc[nt][j] = v;
        s  += v;
        s2 += v * v;
      }
      // butterfly reduce over the 16 lanes holding this row (halves are distinct rows)
#pragma unroll
      for (int off = 1; off < 16; off <<= 1) {
        s  += __shfl_xor(s,  off, 32);
        s2 += __shfl_xor(s2, off, 32);
      }
      const float mu = s * (1.0f / 128.0f);
      mean[j] = mu;
      rstd[j] = rsqrtf(s2 * (1.0f / 128.0f) - mu * mu + 1e-5f);
    }
#pragma unroll
    for (int j = 0; j < 8; ++j) {
      const int m = ((lane >> 4) << 3) + j;
      const size_t rowoff = (size_t)(r0 + m) * D_DIM + (lane & 15);
      float*       dbp = d_bond + rowoff;
      float*       o0p = out0 + rowoff;
      const float* bp  = bond + rowoff;
#pragma unroll
      for (int nt = 0; nt < 8; ++nt) {
        const float y = (acc[nt][j] - mean[j]) * rstd[j] * g[nt] + be[nt];
        dbp[nt * 16] = y;
        o0p[nt * 16] = bp[nt * 16] + y;
      }
    }
  }
}

// ---------------------------------------------------------------------------
// Kernel 3: br = mean_k(coef * d_bond[edge_order]);
//           d_tgt = LN(silu(br@W_e2t + tgt@W_t2t));  out2 = tgt + d_tgt
// ---------------------------------------------------------------------------
__global__ __launch_bounds__(256, 2)
void tgt_kernel(const float* __restrict__ dB,
                const float* __restrict__ tgt,
                const int*   __restrict__ edge_order,
                const float* __restrict__ coef,
                const float* __restrict__ W_e2t,
                const float* __restrict__ W_t2t,
                const float* __restrict__ ln_g,
                const float* __restrict__ ln_b,
                float* __restrict__ out2) {
  __shared__ __attribute__((aligned(32))) __bf16 wT1[D_DIM * D_DIM];
  __shared__ __attribute__((aligned(32))) __bf16 wT2[D_DIM * D_DIM];
  stage_weight_bf16(W_e2t, wT1);
  stage_weight_bf16(W_t2t, wT2);
  __syncthreads();
  const int lane   = threadIdx.x & 31;
  const int wave   = threadIdx.x >> 5;
  const int wpb    = blockDim.x >> 5;
  const int ntiles = (B_SZ * NT_SZ) >> 4;
  float g[8], be[8];
#pragma unroll
  for (int nt = 0; nt < 8; ++nt) {
    g[nt]  = ln_g[nt * 16 + (lane & 15)];
    be[nt] = ln_b[nt * 16 + (lane & 15)];
  }
  for (int tile = blockIdx.x * wpb + wave; tile < ntiles; tile += gridDim.x * wpb) {
    const int r0 = tile << 4;
    const int b  = r0 / NT_SZ;         // NT % 16 == 0, tiles don't cross batches
    const int m  = lane & 15;          // A-layout row owned by this lane
    const int t  = (r0 - b * NT_SZ) + m;
    const float* dbbase = dB + (size_t)b * E_SZ * D_DIM;
    v8f acc[8] = {};
    // ---- Phase 1: bond_reduce fragments (register-built K=16 weighted gather) @ W_e2t
    {
      int   eidx[K_SZ];
      float cf[K_SZ];
#pragma unroll
      for (int k = 0; k < K_SZ; ++k) {
        eidx[k] = edge_order[t * K_SZ + k];
        cf[k]   = coef[t * K_SZ + k];
      }
      v16bf ab[4];
      const int khalf = (lane >> 4) << 3;
#pragma unroll
      for (int kt = 0; kt < 4; ++kt) {
#pragma unroll
        for (int h = 0; h < 2; ++h) {
          const int k0 = kt * 32 + h * 16 + khalf;
          float red[8] = {0.f, 0.f, 0.f, 0.f, 0.f, 0.f, 0.f, 0.f};
#pragma unroll
          for (int k = 0; k < K_SZ; ++k) {
            const float* drow = dbbase + (size_t)eidx[k] * D_DIM + k0;
            const float  c    = cf[k];
#pragma unroll
            for (int j = 0; j < 8; ++j) red[j] += c * drow[j];
          }
#pragma unroll
          for (int j = 0; j < 8; ++j)
            ab[kt][h * 8 + j] = (__bf16)(red[j] * (1.0f / (float)K_SZ));
        }
      }
#pragma unroll
      for (int kt = 0; kt < 4; ++kt) {
        v16bf bw[8];
#pragma unroll
        for (int nt = 0; nt < 8; ++nt) bw[nt] = load_b_frag_lds(wT1, lane, nt, kt);
#pragma unroll
        for (int nt = 0; nt < 8; ++nt)
          acc[nt] = __builtin_amdgcn_wmma_f32_16x16x32_bf16(false, ab[kt], false, bw[nt],
                                                            (short)0, acc[nt], false, false);
      }
    }
    // ---- Phase 2: tgt fragments @ W_t2t
    {
      v16bf at[4];
      const float* trow = tgt + (size_t)(r0 + m) * D_DIM;
#pragma unroll
      for (int kt = 0; kt < 4; ++kt) at[kt] = load_a_frag_global(trow, lane, kt);
#pragma unroll
      for (int kt = 0; kt < 4; ++kt) {
        v16bf bw[8];
#pragma unroll
        for (int nt = 0; nt < 8; ++nt) bw[nt] = load_b_frag_lds(wT2, lane, nt, kt);
#pragma unroll
        for (int nt = 0; nt < 8; ++nt)
          acc[nt] = __builtin_amdgcn_wmma_f32_16x16x32_bf16(false, at[kt], false, bw[nt],
                                                            (short)0, acc[nt], false, false);
      }
    }
    // ---- silu + LN + store
    float mean[8], rstd[8];
#pragma unroll
    for (int j = 0; j < 8; ++j) {
      float s = 0.f, s2 = 0.f;
#pragma unroll
      for (int nt = 0; nt < 8; ++nt) {
        float v = silu_f(acc[nt][j]);
        acc[nt][j] = v;
        s  += v;
        s2 += v * v;
      }
#pragma unroll
      for (int off = 1; off < 16; off <<= 1) {
        s  += __shfl_xor(s,  off, 32);
        s2 += __shfl_xor(s2, off, 32);
      }
      const float mu = s * (1.0f / 128.0f);
      mean[j] = mu;
      rstd[j] = rsqrtf(s2 * (1.0f / 128.0f) - mu * mu + 1e-5f);
    }
#pragma unroll
    for (int j = 0; j < 8; ++j) {
      const int mr = ((lane >> 4) << 3) + j;
      const size_t rowoff = (size_t)(r0 + mr) * D_DIM + (lane & 15);
      float*       o2p = out2 + rowoff;
      const float* tp  = tgt + rowoff;
#pragma unroll
      for (int nt = 0; nt < 8; ++nt) {
        const float y = (acc[nt][j] - mean[j]) * rstd[j] * g[nt] + be[nt];
        o2p[nt * 16] = tp[nt * 16] + y;
      }
    }
  }
}

extern "C" void kernel_launch(void* const* d_in, const int* in_sizes, int n_in,
                              void* d_out, int out_size, void* d_ws, size_t ws_size,
                              hipStream_t stream) {
  (void)in_sizes; (void)n_in; (void)out_size; (void)ws_size;
  const float* bond  = (const float*)d_in[0];
  const float* src   = (const float*)d_in[1];
  const float* tgt   = (const float*)d_in[2];
  const float* W_s2e = (const float*)d_in[3];
  const float* W_t2e = (const float*)d_in[4];
  const float* W_e2e = (const float*)d_in[5];
  const float* ln1_g = (const float*)d_in[6];
  const float* ln1_b = (const float*)d_in[7];
  const float* W_e2t = (const float*)d_in[8];
  const float* W_t2t = (const float*)d_in[9];
  const float* ln2_g = (const float*)d_in[10];
  const float* ln2_b = (const float*)d_in[11];
  const float* coef  = (const float*)d_in[12];
  const int* src_order  = (const int*)d_in[13];
  const int* tgt_order  = (const int*)d_in[14];
  const int* edge_order = (const int*)d_in[15];

  float* out0 = (float*)d_out;                               // [B,E,D]
  float* out1 = out0 + (size_t)B_SZ * E_SZ * D_DIM;          // [B,NS,D]
  float* out2 = out1 + (size_t)B_SZ * NS_SZ * D_DIM;         // [B,NT,D]

  float* Ps = (float*)d_ws;
  float* Pt = Ps + (size_t)B_SZ * NS_SZ * D_DIM;
  float* dB = Pt + (size_t)B_SZ * NT_SZ * D_DIM;

  proj_gemm_kernel<<<320, 256, 0, stream>>>(src, W_s2e, Ps, B_SZ * NS_SZ);
  proj_gemm_kernel<<<320, 256, 0, stream>>>(tgt, W_t2e, Pt, B_SZ * NT_SZ);
  edge_kernel<<<5000, 256, 0, stream>>>(bond, Ps, Pt, src_order, tgt_order,
                                        W_e2e, ln1_g, ln1_b, dB, out0);
  tgt_kernel<<<320, 256, 0, stream>>>(dB, tgt, edge_order, coef,
                                      W_e2t, W_t2t, ln2_g, ln2_b, out2);
  hipMemcpyAsync(out1, src, (size_t)B_SZ * NS_SZ * D_DIM * sizeof(float),
                 hipMemcpyDeviceToDevice, stream);
}
